// GraphDiffusion_66357244723790
// MI455X (gfx1250) — 8-bit matrix operand order repaired and checked by reading, compiled, not run on hardware
//
#include <hip/hip_runtime.h>
#include <stdint.h>

#define N_ROWS  4096
#define D_DIM   256
#define M_LEN   1024
#define N_STEPS 30
#define TEMP_F  0.1f
#define ALPHA_F 0.5f
#define EPS_F   1e-8f

typedef float v8f __attribute__((ext_vector_type(8)));
typedef float v2f __attribute__((ext_vector_type(2)));
typedef int   v8i __attribute__((ext_vector_type(8)));

__global__ __launch_bounds__(256) void k_pack(const float* __restrict__ masks,
                                              uint8_t* __restrict__ mu8,
                                              float* __restrict__ areas) {
  const int row = blockIdx.x, t = threadIdx.x;
  float4 v = ((const float4*)(masks + (size_t)row * M_LEN))[t];
  uint32_t b0 = v.x != 0.0f, b1 = v.y != 0.0f, b2 = v.z != 0.0f, b3 = v.w != 0.0f;
  ((uint32_t*)(mu8 + (size_t)row * M_LEN))[t] = b0 | (b1 << 8) | (b2 << 16) | (b3 << 24);
  __shared__ float red[256];
  red[t] = (float)(b0 + b1 + b2 + b3);
  __syncthreads();
  for (int d = 128; d > 0; d >>= 1) { if (t < d) red[t] += red[t + d]; __syncthreads(); }
  if (t == 0) areas[row] = red[0];
}

__global__ void k_colsum(const uint8_t* __restrict__ mu8, float* __restrict__ colsum) {
  int k = blockIdx.x * blockDim.x + threadIdx.x;
  if (k >= M_LEN) return;
  uint32_t s = 0;
  for (int i = 0; i < N_ROWS; ++i) s += mu8[(size_t)i * M_LEN + k];
  colsum[k] = (float)s;
}

__global__ __launch_bounds__(256) void k_coef(const uint8_t* __restrict__ mu8,
                                              const float* __restrict__ colsum,
                                              const float* __restrict__ areas,
                                              float* __restrict__ coef,
                                              float* __restrict__ validf) {
  const int row = blockIdx.x, t = threadIdx.x;
  float s = 0.f;
  for (int k = t; k < M_LEN; k += 256) s += colsum[k] * (float)mu8[(size_t)row * M_LEN + k];
  __shared__ float red[256];
  red[t] = s;
  __syncthreads();
  for (int d = 128; d > 0; d >>= 1) { if (t < d) red[t] += red[t + d]; __syncthreads(); }
  if (t == 0) {
    float ae    = areas[row] + EPS_F;
    float wsum  = red[0] / ae;
    int   valid = wsum > 0.f;
    float sw    = valid ? wsum : 1.f;
    coef[row]   = 1.f / (ae * TEMP_F * sw);
    validf[row] = valid ? 1.f : 0.f;
  }
}

__device__ __forceinline__ int offA8(int i, int hf) {
  return 32 * (i >> 2) + 16 * ((i >> 1) & 1) + 8 * hf + 4 * (i & 1);
}
__device__ __forceinline__ int offB8(int i, int hf) {
  return 32 * (i >> 2) + 16 * ((i >> 1) & 1) + 8 * hf + 4 * (i & 1);
}

__global__ __launch_bounds__(256) void k_gemm_s(const float* __restrict__ F,
                                                const uint8_t* __restrict__ mu8,
                                                const float* __restrict__ coef,
                                                float* __restrict__ S) {
  const int wave = threadIdx.x >> 5, lane = threadIdx.x & 31;
  const int hf = lane >> 4, q = lane & 15;
  const int row0 = blockIdx.y * 128 + wave * 16;
  const int col0 = blockIdx.x * 16;

  const uint8_t* arow = mu8 + (size_t)(row0 + q) * M_LEN;
  const uint8_t* brow = mu8 + (size_t)(col0 + q) * M_LEN;
  v8i ci = {};
  for (int kb = 0; kb < M_LEN; kb += 64) {
    v8i ai{}, bi{};
#pragma unroll
    for (int i = 0; i < 8; ++i) {
      ai[i] = (int)*(const uint32_t*)(arow + kb + offA8(i, hf));
      bi[i] = (int)*(const uint32_t*)(brow + kb + offB8(i, hf));
    }
    ci = __builtin_amdgcn_wmma_i32_16x16x64_iu8(false, ai, false, bi, ci, false, false);
  }

  const float* fa = F + (size_t)(row0 + q) * D_DIM;
  const float* fb = F + (size_t)(col0 + q) * D_DIM;
  v8f cf = {};
  for (int k0 = 0; k0 < D_DIM; k0 += 4) {
    v2f a = *(const v2f*)(fa + k0 + 2 * hf);
    v2f b = *(const v2f*)(fb + k0 + 2 * hf);
    cf = __builtin_amdgcn_wmma_f32_16x16x4_f32(false, a, false, b, (short)0, cf,
                                               false, false);
  }

#pragma unroll
  for (int r = 0; r < 8; ++r) {
    int gi = row0 + r + 8 * hf;
    S[(size_t)gi * N_ROWS + col0 + q] = cf[r] * (float)ci[r] * coef[gi];
  }
}

__global__ __launch_bounds__(256) void k_softmax(float* __restrict__ S,
                                                 const float* __restrict__ validf) {
  const int row = blockIdx.x, t = threadIdx.x;
  float* Sr = S + (size_t)row * N_ROWS;
  __shared__ float red[256];
  __shared__ float s_max, s_scale;

  float m = -3.4e38f;
  for (int j = t; j < N_ROWS; j += 256) m = fmaxf(m, Sr[j]);
  red[t] = m;
  __syncthreads();
  for (int d = 128; d > 0; d >>= 1) { if (t < d) red[t] = fmaxf(red[t], red[t + d]); __syncthreads(); }
  if (t == 0) s_max = red[0];
  __syncthreads();
  const float mm = s_max;

  float s = 0.f;
  for (int j = t; j < N_ROWS; j += 256) s += __expf(Sr[j] - mm);
  red[t] = s;
  __syncthreads();
  for (int d = 128; d > 0; d >>= 1) { if (t < d) red[t] += red[t + d]; __syncthreads(); }
  if (t == 0) s_scale = validf[row] / red[0];
  __syncthreads();
  const float sc = s_scale;

  for (int j = t; j < N_ROWS; j += 256) Sr[j] = __expf(Sr[j] - mm) * sc;
}

__global__ __launch_bounds__(256) void k_matvec(const float* __restrict__ W,
                                                const float* __restrict__ rin,
                                                const float* __restrict__ s0,
                                                float* __restrict__ rout) {
  __shared__ __align__(16) float sh[N_ROWS];
  const int t = threadIdx.x;
  for (int j = t; j < N_ROWS; j += 256) sh[j] = rin[j];
  __syncthreads();

  const int wave = t >> 5, lane = t & 31;
  const int row = blockIdx.x * 8 + wave;
  const float4* Wr = (const float4*)(W + (size_t)row * N_ROWS);
  const float4* Rv = (const float4*)sh;
  float acc = 0.f;
#pragma unroll 4
  for (int c = 0; c < N_ROWS / 128; ++c) {
    int j4 = c * 32 + lane;
    float4 w = Wr[j4];
    float4 r = Rv[j4];
    acc += w.x * r.x + w.y * r.y + w.z * r.z + w.w * r.w;
  }
  for (int off = 16; off > 0; off >>= 1) acc += __shfl_down(acc, off, 32);
  if (lane == 0) rout[row] = ALPHA_F * acc + (1.f - ALPHA_F) * s0[row];
}

extern "C" void kernel_launch(void* const* d_in, const int* in_sizes, int n_in,
                              void* d_out, int out_size, void* d_ws, size_t ws_size,
                              hipStream_t stream) {
  (void)in_sizes; (void)n_in; (void)out_size; (void)ws_size;
  const float* features = (const float*)d_in[0];
  const float* masks    = (const float*)d_in[1];
  const float* s0       = (const float*)d_in[2];

  char* ws = (char*)d_ws;
  size_t off = 0;
  float*   S      = (float*)(ws + off); off += (size_t)N_ROWS * N_ROWS * sizeof(float);
  uint8_t* mu8    = (uint8_t*)(ws + off); off += (size_t)N_ROWS * M_LEN;
  float*   areas  = (float*)(ws + off); off += N_ROWS * sizeof(float);
  float*   colsum = (float*)(ws + off); off += M_LEN * sizeof(float);
  float*   coef   = (float*)(ws + off); off += N_ROWS * sizeof(float);
  float*   validf = (float*)(ws + off); off += N_ROWS * sizeof(float);
  float*   rA     = (float*)(ws + off); off += N_ROWS * sizeof(float);
  float*   rB     = (float*)(ws + off); off += N_ROWS * sizeof(float);

  k_pack<<<N_ROWS, 256, 0, stream>>>(masks, mu8, areas);
  k_colsum<<<M_LEN / 256, 256, 0, stream>>>(mu8, colsum);
  k_coef<<<N_ROWS, 256, 0, stream>>>(mu8, colsum, areas, coef, validf);
  k_gemm_s<<<dim3(N_ROWS / 16, N_ROWS / 128), 256, 0, stream>>>(features, mu8, coef, S);
  k_softmax<<<N_ROWS, 256, 0, stream>>>(S, validf);

  const float* src = s0;
  for (int step = 0; step < N_STEPS; ++step) {
    float* dst = (step == N_STEPS - 1) ? (float*)d_out : ((step & 1) ? rB : rA);
    k_matvec<<<N_ROWS / 8, 256, 0, stream>>>(S, src, s0, dst);
    src = dst;
  }
}
